// HTN_68135361183826
// MI455X (gfx1250) — matrix operand order repaired and checked by reading, not compiled, not run on hardware
//
#include <hip/hip_runtime.h>
#include <hip/hip_bf16.h>
#include <hip/hip_fp16.h>

// ---------------------------------------------------------------------------
// CDNA5 (gfx1250) HTN: wave32 WMMA f16 16x16x32 for all GEMM-shaped work,
// async global->LDS staging (ASYNCcnt) for pre-transposed f16 weights.
// ---------------------------------------------------------------------------

typedef _Float16 v16h __attribute__((ext_vector_type(16)));
typedef _Float16 h8   __attribute__((ext_vector_type(8)));
typedef float    v8f  __attribute__((ext_vector_type(8)));

#define DEVINL __device__ __forceinline__

// Wave-local LDS drain (CDNA5 split counters).
DEVINL void lds_fence() { asm volatile("s_wait_dscnt 0" ::: "memory"); }

// Async global->LDS copy, 16B per lane. Tracked by ASYNCcnt.
DEVINL void async_copy_b128(unsigned int lds_byte_addr, const void* gptr) {
  asm volatile("global_load_async_to_lds_b128 %0, %1, off"
               :: "v"(lds_byte_addr), "v"((unsigned long long)(uintptr_t)gptr)
               : "memory");
}
DEVINL void wait_async() { asm volatile("s_wait_asynccnt 0x0" ::: "memory"); }

DEVINL v8f wmma16(v16h a, v16h b, v8f c) {
  return __builtin_amdgcn_wmma_f32_16x16x32_f16(
      /*neg_a=*/false, a, /*neg_b=*/false, b,
      /*c_mod=*/(short)0, c, /*reuse_a=*/false, /*reuse_b=*/false);
}

// A fragment (16x32 f16, M=lane%16): lane half=lane/16 needs halves
// [k0+8h, +8) and [k0+16+8h, +8) of its row -> two 16B loads.
DEVINL v16h afrag(const _Float16* rowp, int k0, int half) {
  h8 lo = *(const h8*)(rowp + k0 + 8 * half);
  h8 hi = *(const h8*)(rowp + k0 + 16 + 8 * half);
  v16h a;
#pragma unroll
  for (int i = 0; i < 8; ++i) { a[i] = lo[i]; a[i + 8] = hi[i]; }
  return a;
}

// B fragment (32x16 f16, N=lane%16) from W^T stored [64+ out][K]:
// element j<8 -> K = k0 + 8*half + j, j>=8 -> K = k0 + 16 + 8*half + (j-8).
DEVINL v16h bfrag(const _Float16* wt, int K, int ncol, int k0, int half) {
  const _Float16* p = wt + (size_t)ncol * K + k0 + 8 * half;
  h8 lo = *(const h8*)p;
  h8 hi = *(const h8*)(p + 16);
  v16h b;
#pragma unroll
  for (int i = 0; i < 8; ++i) { b[i] = lo[i]; b[i + 8] = hi[i]; }
  return b;
}

// ---------------------------------------------------------------------------
// Prep: f32 -> f16 transpose (dst[Nout][K] = src[K][Nout]) and plain convert.
// ---------------------------------------------------------------------------
__global__ void htn_transpose_f16(const float* __restrict__ src,
                                  _Float16* __restrict__ dst, int K, int Nout) {
  int idx = blockIdx.x * blockDim.x + threadIdx.x;
  if (idx >= K * Nout) return;
  int n = idx / K, k = idx % K;
  dst[idx] = (_Float16)src[(size_t)k * Nout + n];
}

__global__ void htn_cvt_f16(const float* __restrict__ src,
                            _Float16* __restrict__ dst, int count) {
  int idx = blockIdx.x * blockDim.x + threadIdx.x;
  if (idx < count) dst[idx] = (_Float16)src[idx];
}

// ---------------------------------------------------------------------------
// Projection: proj = X[n,KIN] @ Wp -> [n,256]. A read directly from global f16;
// Wp^T (pre-transposed f16) async-staged into LDS in 64-deep K chunks.
// Block = 256 (8 waves): 32 rows x 256 cols per block.
// ---------------------------------------------------------------------------
template <int KIN>
__global__ __launch_bounds__(256) void htn_proj_kernel(
    const _Float16* __restrict__ Xh, const _Float16* __restrict__ WpT, // [256][KIN]
    float* __restrict__ proj, _Float16* __restrict__ projh, int nrows) {
  __shared__ __align__(16) _Float16 sBT[256 * 64];   // Wp^T chunk [col][k], 32KB

  const int tid = threadIdx.x;
  const int R0 = blockIdx.x * 32;
  const int w = tid >> 5, lane = tid & 31;
  const int rowgrp = w >> 2, wquad = w & 3;
  const int half = lane >> 4, nloc = lane & 15;

  int arown = R0 + rowgrp * 16 + nloc;
  if (arown >= nrows) arown = nrows - 1;
  const _Float16* arow = Xh + (size_t)arown * KIN;

  v8f acc[4] = {};
  for (int kc = 0; kc < KIN / 64; ++kc) {
    __syncthreads();  // prior chunk readers done before overwrite
#pragma unroll
    for (int j = 0; j < 8; ++j) {
      int idx = tid + j * 256;                 // 2048 x b128 transfers
      int c = idx >> 3, off8 = (idx & 7) * 8;
      async_copy_b128((unsigned int)(uintptr_t)&sBT[idx * 8],
                      WpT + (size_t)c * KIN + kc * 64 + off8);
    }
    wait_async();
    __syncthreads();
#pragma unroll
    for (int c2 = 0; c2 < 2; ++c2) {
      v16h A = afrag(arow, kc * 64 + c2 * 32, half);
#pragma unroll
      for (int t = 0; t < 4; ++t)
        acc[t] = wmma16(A, bfrag(sBT, 64, wquad * 64 + t * 16 + nloc, c2 * 32, half), acc[t]);
    }
  }

#pragma unroll
  for (int t = 0; t < 4; ++t) {
    int col = wquad * 64 + t * 16 + nloc;
#pragma unroll
    for (int r = 0; r < 8; ++r) {
      int row = R0 + rowgrp * 16 + r + 8 * half;
      if (row < nrows) {
        float v = acc[t][r];
        proj[(size_t)row * 256 + col] = v;
        projh[(size_t)row * 256 + col] = (_Float16)v;
      }
    }
  }
}

// ---------------------------------------------------------------------------
// Edge kernel: 4 waves/block, each wave loops 4 groups of (4 edges x 4 heads)
// = M=16 WMMA tiles. Weights are one flat pre-transposed f16 block
// [aW1T(64x192) | eW1T(64x128) | eW2T(64x64)] async-staged once per block.
// ---------------------------------------------------------------------------
__global__ __launch_bounds__(128) void htn_edge_kernel(
    const _Float16* __restrict__ projh,   // [n][256]
    const long long* __restrict__ ei,     // [3][E]
    const _Float16* __restrict__ wT,      // 24576 halves, see layout above
    const float* __restrict__ ab1, const float* __restrict__ aW2,
    const float* __restrict__ ab2, const float* __restrict__ eb1,
    const float* __restrict__ eb2,
    float* __restrict__ ws,               // [3][n][256]
    int E, int n) {
  __shared__ __align__(16) _Float16 s_w[24576];        // 48KB weights
  __shared__ __align__(16) _Float16 s_hid[4][16 * 64]; // 8KB, per-wave staging
  __shared__ float s_sc[4][16];
  __shared__ float s_al[4][16];
  __shared__ int   s_seg[4][3][4];

  const int tid = threadIdx.x;
  for (int idx = tid; idx < 3072; idx += 128)          // 3072 x b128
    async_copy_b128((unsigned int)(uintptr_t)&s_w[idx * 8], wT + (size_t)idx * 8);
  wait_async();
  __syncthreads();
  const _Float16* s_aW1T = s_w;            // [64][192]
  const _Float16* s_eW1T = s_w + 12288;    // [64][128]
  const _Float16* s_eW2T = s_w + 20480;    // [64][64]

  const int w = tid >> 5, lane = tid & 31;
  const int half = lane >> 4, nloc = lane & 15;
  const int myrow = nloc;
  const float ab2v = ab2[0];
  const int g0 = (blockIdx.x * 4 + w) * 4;             // 4 groups per wave

#pragma unroll 1
  for (int g = 0; g < 4; ++g) {
    const int ebase = (g0 + g) * 4;
    int e = ebase + (myrow >> 2); if (e >= E) e = E - 1;
    const int head = myrow & 3;

    const long long i0 = ei[0 * (long long)E + e];
    const long long i1 = ei[1 * (long long)E + e];
    const long long i2 = ei[2 * (long long)E + e];
    const _Float16* dptr = projh + (size_t)i0 * 256 + head * 64;
    const _Float16* mptr = projh + (size_t)i1 * 256 + head * 64;
    const _Float16* sptr = projh + (size_t)i2 * 256 + head * 64;

    if (lane < 16) {
      s_sc[w][lane] = 0.f;                 // DS ops are wave-in-order: safe
      if ((lane & 3) == 0) {
        int eg = lane >> 2;
        s_seg[w][0][eg] = (int)i0;
        s_seg[w][1][eg] = (int)i1;
        s_seg[w][2][eg] = (int)i2;
      }
    }
    lds_fence();

    // ---- attention: [16,192] @ aW1 -> relu -> @ aW2 -> sc[16] ----
    v8f acc[4] = {};
#pragma unroll
    for (int c = 0; c < 6; ++c) {
      const _Float16* src = (c < 2) ? dptr : (c < 4) ? mptr : sptr;
      v16h A = afrag(src, (c & 1) * 32, half);
#pragma unroll
      for (int t = 0; t < 4; ++t)
        acc[t] = wmma16(A, bfrag(s_aW1T, 192, t * 16 + nloc, c * 32, half), acc[t]);
    }
#pragma unroll
    for (int t = 0; t < 4; ++t) {
      int col = t * 16 + nloc;
      float w2 = aW2[col], b1 = ab1[col];
#pragma unroll
      for (int r = 0; r < 8; ++r) {
        float h = acc[t][r] + b1;
        h = h > 0.f ? h : 0.f;
        atomicAdd(&s_sc[w][r + 8 * half], h * w2);     // ds_add_f32
      }
    }
    lds_fence();

    // ---- leaky + softmax over the 4 heads of each edge ----
    if (lane < 16) {
      int gb = lane & ~3;
      float v0 = s_sc[w][gb + 0] + ab2v; v0 = v0 > 0.f ? v0 : 0.2f * v0;
      float v1 = s_sc[w][gb + 1] + ab2v; v1 = v1 > 0.f ? v1 : 0.2f * v1;
      float v2 = s_sc[w][gb + 2] + ab2v; v2 = v2 > 0.f ? v2 : 0.2f * v2;
      float v3 = s_sc[w][gb + 3] + ab2v; v3 = v3 > 0.f ? v3 : 0.2f * v3;
      float mx = fmaxf(fmaxf(v0, v1), fmaxf(v2, v3));
      float e0 = __expf(v0 - mx), e1 = __expf(v1 - mx);
      float e2 = __expf(v2 - mx), e3 = __expf(v3 - mx);
      float den = e0 + e1 + e2 + e3;
      int hsel = lane & 3;
      float num = hsel == 0 ? e0 : hsel == 1 ? e1 : hsel == 2 ? e2 : e3;
      s_al[w][lane] = num / den;
    }
    lds_fence();

    // ---- 3 neighbor MLPs + alpha-weighted atomic segment-sum ----
#pragma unroll 1
    for (int i = 0; i < 3; ++i) {
      const _Float16* pa = (i == 0) ? mptr : dptr;
      const _Float16* pb = (i == 2) ? mptr : sptr;

      v8f h1[4] = {};
#pragma unroll
      for (int c = 0; c < 4; ++c) {
        const _Float16* src = (c < 2) ? pa : pb;
        v16h A = afrag(src, (c & 1) * 32, half);
#pragma unroll
        for (int t = 0; t < 4; ++t)
          h1[t] = wmma16(A, bfrag(s_eW1T, 128, t * 16 + nloc, c * 32, half), h1[t]);
      }
#pragma unroll
      for (int t = 0; t < 4; ++t) {
        int col = t * 16 + nloc;
        float b1 = eb1[col];
#pragma unroll
        for (int r = 0; r < 8; ++r) {
          float v = h1[t][r] + b1;
          v = v > 0.f ? v : 0.2f * v;                  // leaky_relu(0.2)
          s_hid[w][(r + 8 * half) * 64 + col] = (_Float16)v;
        }
      }
      lds_fence();

      v8f h2[4] = {};
      const _Float16* hrow = &s_hid[w][myrow * 64];
#pragma unroll
      for (int c = 0; c < 2; ++c) {
        v16h A = afrag(hrow, c * 32, half);
#pragma unroll
        for (int t = 0; t < 4; ++t)
          h2[t] = wmma16(A, bfrag(s_eW2T, 64, t * 16 + nloc, c * 32, half), h2[t]);
      }
#pragma unroll
      for (int t = 0; t < 4; ++t) {
        int col = t * 16 + nloc;
        float b2 = eb2[col];
#pragma unroll
        for (int r = 0; r < 8; ++r) {
          int M = r + 8 * half;
          float v = (h2[t][r] + b2) * s_al[w][M];
          int seg = s_seg[w][i][M >> 2];
          unsafeAtomicAdd(ws + (((size_t)i * n + seg) * 256 + (M & 3) * 64 + col), v);
        }
      }
    }
  }
}

// ---------------------------------------------------------------------------
// Layer-0 epilogue: out = sigmoid(theta*proj + ws + bias); write f32 + f16.
// ---------------------------------------------------------------------------
__global__ __launch_bounds__(256) void htn_combine0_kernel(
    const float* __restrict__ proj, const float* __restrict__ ws,
    const float* __restrict__ theta, const float* __restrict__ bias,
    float* __restrict__ x1, _Float16* __restrict__ x1h, int n) {
  size_t idx = (size_t)blockIdx.x * blockDim.x + threadIdx.x;
  if (idx >= (size_t)n * 256) return;
  int c = (int)(idx & 255);
  float base = theta[c] * proj[idx] + bias[c];
#pragma unroll
  for (int i = 0; i < 3; ++i) {
    size_t oi = (size_t)i * n * 256 + idx;
    float o = base + ws[oi];
    o = 1.f / (1.f + __expf(-o));
    x1[oi] = o;
    x1h[oi] = (_Float16)o;
  }
}

// ---------------------------------------------------------------------------
// Layer-1 epilogue: mean over heads + bias (no activation).
// ---------------------------------------------------------------------------
__global__ __launch_bounds__(256) void htn_combine1_kernel(
    const float* __restrict__ proj, const float* __restrict__ ws,
    const float* __restrict__ theta, const float* __restrict__ bias,
    float* __restrict__ out, int n) {
  size_t idx = (size_t)blockIdx.x * blockDim.x + threadIdx.x;
  if (idx >= (size_t)n * 64) return;
  int f = (int)(idx & 63);
  size_t v = idx >> 6;
  float tp = 0.f;
#pragma unroll
  for (int h = 0; h < 4; ++h) tp += theta[h * 64 + f] * proj[v * 256 + h * 64 + f];
  float b = bias[f];
#pragma unroll
  for (int i = 0; i < 3; ++i) {
    float s = tp;
#pragma unroll
    for (int h = 0; h < 4; ++h) s += ws[(size_t)i * n * 256 + v * 256 + h * 64 + f];
    out[((size_t)i * n + v) * 64 + f] = s * 0.25f + b;
  }
}

// ---------------------------------------------------------------------------
extern "C" void kernel_launch(void* const* d_in, const int* in_sizes, int n_in,
                              void* d_out, int out_size, void* d_ws, size_t ws_size,
                              hipStream_t stream) {
  (void)in_sizes; (void)n_in; (void)out_size; (void)ws_size;
  const int N = 20000, E = 200000;

  const float*     x0 = (const float*)d_in[0];
  const long long* ei = (const long long*)d_in[1];   // int64 edge_index [3][E]
  const float* p0_Wp  = (const float*)d_in[2];
  const float* p0_aW1 = (const float*)d_in[3];
  const float* p0_ab1 = (const float*)d_in[4];
  const float* p0_aW2 = (const float*)d_in[5];
  const float* p0_ab2 = (const float*)d_in[6];
  const float* p0_eW1 = (const float*)d_in[7];
  const float* p0_eb1 = (const float*)d_in[8];
  const float* p0_eW2 = (const float*)d_in[9];
  const float* p0_eb2 = (const float*)d_in[10];
  const float* p0_th  = (const float*)d_in[11];
  const float* p0_b   = (const float*)d_in[12];
  const float* p1_Wp  = (const float*)d_in[13];
  const float* p1_aW1 = (const float*)d_in[14];
  const float* p1_ab1 = (const float*)d_in[15];
  const float* p1_aW2 = (const float*)d_in[16];
  const float* p1_ab2 = (const float*)d_in[17];
  const float* p1_eW1 = (const float*)d_in[18];
  const float* p1_eb1 = (const float*)d_in[19];
  const float* p1_eW2 = (const float*)d_in[20];
  const float* p1_eb2 = (const float*)d_in[21];
  const float* p1_th  = (const float*)d_in[22];
  const float* p1_b   = (const float*)d_in[23];
  float* out = (float*)d_out;

  char* wsb = (char*)d_ws;
  size_t off = 0;
  auto carve = [&](size_t bytes) -> void* {
    void* p = wsb + off;
    off = (off + bytes + 255) & ~(size_t)255;
    return p;
  };
  float*    proj0  = (float*)carve((size_t)N * 256 * 4);
  _Float16* projh0 = (_Float16*)carve((size_t)N * 256 * 2);
  float*    ws0    = (float*)carve((size_t)3 * N * 256 * 4);
  float*    x1     = (float*)carve((size_t)3 * N * 256 * 4);
  _Float16* x1h    = (_Float16*)carve((size_t)3 * N * 256 * 2);
  float*    proj1  = (float*)carve((size_t)3 * N * 256 * 4);
  _Float16* projh1 = (_Float16*)carve((size_t)3 * N * 256 * 2);
  float*    ws1    = (float*)carve((size_t)9 * N * 256 * 4);
  _Float16* x0h    = (_Float16*)carve((size_t)N * 64 * 2);
  _Float16* WpT0   = (_Float16*)carve((size_t)256 * 64 * 2);
  _Float16* WpT1   = (_Float16*)carve((size_t)256 * 256 * 2);
  _Float16* wT0    = (_Float16*)carve((size_t)24576 * 2);
  _Float16* wT1    = (_Float16*)carve((size_t)24576 * 2);

  hipMemsetAsync(ws0, 0, (size_t)3 * N * 256 * 4, stream);
  hipMemsetAsync(ws1, 0, (size_t)9 * N * 256 * 4, stream);

  // ---- one-time prep: f16 transposed weights + f16 input copy ----
  auto tgrid = [](int elems) { return (elems + 255) / 256; };
  htn_cvt_f16<<<tgrid(N * 64), 256, 0, stream>>>(x0, x0h, N * 64);
  htn_transpose_f16<<<tgrid(64 * 256), 256, 0, stream>>>(p0_Wp, WpT0, 64, 256);
  htn_transpose_f16<<<tgrid(256 * 256), 256, 0, stream>>>(p1_Wp, WpT1, 256, 256);
  htn_transpose_f16<<<tgrid(192 * 64), 256, 0, stream>>>(p0_aW1, wT0, 192, 64);
  htn_transpose_f16<<<tgrid(128 * 64), 256, 0, stream>>>(p0_eW1, wT0 + 12288, 128, 64);
  htn_transpose_f16<<<tgrid(64 * 64),  256, 0, stream>>>(p0_eW2, wT0 + 20480, 64, 64);
  htn_transpose_f16<<<tgrid(192 * 64), 256, 0, stream>>>(p1_aW1, wT1, 192, 64);
  htn_transpose_f16<<<tgrid(128 * 64), 256, 0, stream>>>(p1_eW1, wT1 + 12288, 128, 64);
  htn_transpose_f16<<<tgrid(64 * 64),  256, 0, stream>>>(p1_eW2, wT1 + 20480, 64, 64);

  // ---- layer 0 ----
  htn_proj_kernel<64><<<N / 32, 256, 0, stream>>>(x0h, WpT0, proj0, projh0, N);
  htn_edge_kernel<<<E / 64, 128, 0, stream>>>(projh0, ei, wT0,
      p0_ab1, p0_aW2, p0_ab2, p0_eb1, p0_eb2, ws0, E, N);
  htn_combine0_kernel<<<(N * 256) / 256, 256, 0, stream>>>(proj0, ws0, p0_th, p0_b, x1, x1h, N);

  // ---- layer 1 ----
  htn_proj_kernel<256><<<(3 * N) / 32, 256, 0, stream>>>(x1h, WpT1, proj1, projh1, 3 * N);
  htn_edge_kernel<<<E / 64, 128, 0, stream>>>(projh1, ei, wT1,
      p1_ab1, p1_aW2, p1_ab2, p1_eb1, p1_eb2, ws1, E, 3 * N);
  htn_combine1_kernel<<<(3 * N * 64) / 256, 256, 0, stream>>>(proj1, ws1, p1_th, p1_b, out, 3 * N);
}
